// TransformerBlockQuantum_65481071402936
// MI455X (gfx1250) — matrix operand order repaired and checked by reading, not compiled, not run on hardware
//
#include <hip/hip_runtime.h>

typedef __attribute__((ext_vector_type(16))) __bf16 v16bf;
typedef __attribute__((ext_vector_type(8)))  __bf16 v8bf;
typedef __attribute__((ext_vector_type(8)))  float  v8f;

#define EMBED 1024
#define NQ    32
#define FFN   4096
#define SEQ   1024
#define BATCH 8
#define BS    (BATCH * SEQ)        // 8192 tokens
#define LN_EPS 1e-5f
#define INV_SQRT_DK 0.70710678f    // 1/sqrt(2)

// round-to-nearest-even float -> bf16 (as raw u16)
static __device__ __forceinline__ unsigned short f2bfu(float f) {
    union { float f; unsigned u; } v; v.f = f;
    unsigned r = v.u + 0x7FFFu + ((v.u >> 16) & 1u);
    return (unsigned short)(r >> 16);
}

// ---------------------------------------------------------------------------
// qv = cos(theta_attn[j]) * cos(x[t, j]),  [BS, 32]
// ---------------------------------------------------------------------------
__global__ __launch_bounds__(256) void k_qv(const float* __restrict__ x,
                                            const float* __restrict__ theta,
                                            float* __restrict__ qv) {
    int idx = blockIdx.x * 256 + threadIdx.x;
    if (idx >= BS * NQ) return;
    int row = idx >> 5, j = idx & 31;
    qv[idx] = __cosf(theta[j]) * __cosf(x[(size_t)row * EMBED + j]);
}

// ---------------------------------------------------------------------------
// Pre-swizzle w_ffn2 [1024, 4096] f32 into bf16 WMMA B-fragment order.
// Fragment (kb, ntile) is 1KB: lane l (32B contiguous) holds
//   e = ntile*16 + (l&15),  k = kb*32 + (l>>4)*16 + hw,  hw = 0..15
// ---------------------------------------------------------------------------
__global__ __launch_bounds__(256) void k_prep_w2(const float* __restrict__ w2,
                                                 unsigned short* __restrict__ w2f) {
    int idx = blockIdx.x * 256 + threadIdx.x;
    if (idx >= EMBED * FFN) return;
    int hw    = idx & 15;
    int lane  = (idx >> 4) & 31;
    int frag  = idx >> 9;
    int ntile = frag & 63;
    int kb    = frag >> 6;
    int e = ntile * 16 + (lane & 15);
    int k = kb * 32 + ((lane >> 4) << 4) + hw;
    w2f[idx] = f2bfu(w2[(size_t)e * FFN + k]);
}

// ---------------------------------------------------------------------------
// Flash-style quantum attention, dk = 2 per head (q = k = v).
// grid (4 s-chunks, 128 bh), 256 thr. All 1024 K/V rows live in LDS.
// |score| <= 2/sqrt(2) = 1.42, so raw exp is numerically safe (no max pass).
// ---------------------------------------------------------------------------
__global__ __launch_bounds__(256) void k_attn(const float* __restrict__ qv,
                                              float* __restrict__ attn) {
    int bh = blockIdx.y;
    int b = bh >> 4, h = bh & 15;
    __shared__ float qa[SEQ], qc[SEQ];
    for (int i = threadIdx.x; i < SEQ; i += 256) {
        size_t base = ((size_t)b * SEQ + i) * NQ + 2 * h;
        qa[i] = qv[base];
        qc[i] = qv[base + 1];
    }
    __syncthreads();
    int s = blockIdx.x * 256 + threadIdx.x;
    float a_s = qa[s], c_s = qc[s];
    float l = 0.f, accA = 0.f, accC = 0.f;
    for (int t = 0; t < SEQ; ++t) {
        float sc = (a_s * qa[t] + c_s * qc[t]) * INV_SQRT_DK;
        float p = __expf(sc);
        l += p;
        accA = fmaf(p, qa[t], accA);
        accC = fmaf(p, qc[t], accC);
    }
    float inv = 1.0f / l;
    size_t o = ((size_t)b * SEQ + s) * NQ + 2 * h;
    attn[o]     = accA * inv;
    attn[o + 1] = accC * inv;
}

// ---------------------------------------------------------------------------
// x1 = layernorm(x + attn @ w_combine^T, g1, b1).  One token per block.
// ---------------------------------------------------------------------------
__global__ __launch_bounds__(256) void k_x1(const float* __restrict__ x,
                                            const float* __restrict__ attn,
                                            const float* __restrict__ wc,
                                            const float* __restrict__ g1,
                                            const float* __restrict__ b1,
                                            float* __restrict__ x1) {
    int tok = blockIdx.x;
    int tid = threadIdx.x;
    __shared__ float arow[NQ];
    __shared__ float yrow[EMBED];
    __shared__ float red[2];
    if (tid < NQ) arow[tid] = attn[(size_t)tok * NQ + tid];
    if (tid < 2)  red[tid] = 0.f;
    __syncthreads();
    float lsum = 0.f, lss = 0.f;
    #pragma unroll
    for (int i = 0; i < 4; ++i) {
        int e = tid + i * 256;
        float s = x[(size_t)tok * EMBED + e];
        const float* wr = wc + (size_t)e * NQ;
        #pragma unroll
        for (int j = 0; j < NQ; ++j) s = fmaf(arow[j], wr[j], s);
        yrow[e] = s;
        lsum += s;
        lss  = fmaf(s, s, lss);
    }
    atomicAdd(&red[0], lsum);
    atomicAdd(&red[1], lss);
    __syncthreads();
    float mu = red[0] * (1.0f / EMBED);
    float var = red[1] * (1.0f / EMBED) - mu * mu;
    float rs = rsqrtf(var + LN_EPS);
    #pragma unroll
    for (int i = 0; i < 4; ++i) {
        int e = tid + i * 256;
        x1[(size_t)tok * EMBED + e] = (yrow[e] - mu) * rs * g1[e] + b1[e];
    }
}

// ---------------------------------------------------------------------------
// Fused FFN: qf = cos(tf)*cos(x1[:, :32]); h = relu(qf @ w1^T) (bf16);
// ffn = h @ w2^T via v_wmma_f32_16x16x32_bf16; out = LN(x1 + ffn, g2, b2).
// 16 tokens/block, 8 waves; wave w owns E columns [w*128, w*128+128).
// Pipeline: B loads issued at iteration top (covered by GEMM1 VALU + barrier),
// h tile double-buffered -> single barrier per k-block.
// ---------------------------------------------------------------------------
__global__ __launch_bounds__(256) void k_ffn(const float* __restrict__ x1,
                                             const float* __restrict__ tf,
                                             const float* __restrict__ w1,
                                             const unsigned short* __restrict__ w2f,
                                             const float* __restrict__ g2,
                                             const float* __restrict__ b2,
                                             float* __restrict__ out) {
    const int tid  = threadIdx.x;
    const int lane = tid & 31;
    const int wave = tid >> 5;
    const int half = lane >> 4;   // 0: lanes 0-15, 1: lanes 16-31
    const int ln   = lane & 15;
    const int tok_base = blockIdx.x * 16;

    __shared__ float qf[16 * NQ];                                // 2 KB
    __shared__ __align__(16) unsigned short h_lds[2][16 * 32];   // 2 KB bf16
    __shared__ float rowsum[16], rowss[16], mu_s[16], rs_s[16];

    // qf tile (16 x 32)
    for (int i = tid; i < 16 * NQ; i += 256) {
        int m = i >> 5, j = i & 31;
        qf[i] = __cosf(tf[j]) * __cosf(x1[(size_t)(tok_base + m) * EMBED + j]);
    }
    if (tid < 16) { rowsum[tid] = 0.f; rowss[tid] = 0.f; }
    __syncthreads();

    const int m_h = tid >> 4;           // 0..15  (row for h compute)
    const int kk0 = (tid & 15) * 2;     // 0..30  (2 k-cols per thread)

    // hoist loop-invariant qf row into registers (kills per-iter ds loads)
    float qreg[NQ];
    #pragma unroll
    for (int j = 0; j < NQ; ++j) qreg[j] = qf[m_h * NQ + j];

    v8f acc[8];
    #pragma unroll
    for (int t = 0; t < 8; ++t)
        #pragma unroll
        for (int i = 0; i < 8; ++i) acc[t][i] = 0.f;

    for (int kb = 0; kb < FFN / 32; ++kb) {
        const int buf = kb & 1;
        // ---- B fragments: issue early; GEMM1 + barrier cover the latency ----
        const unsigned short* bbase =
            w2f + ((size_t)(kb * 64 + wave * 8) << 9) + lane * 16;
        v16bf bfr[8];
        #pragma unroll
        for (int t = 0; t < 8; ++t)
            bfr[t] = *(const v16bf*)(bbase + (t << 9));
        if (kb + 1 < FFN / 32)
            __builtin_prefetch(bbase + (64 << 9), 0, 1);
        // ---- GEMM1 + relu: h[16 x 32] for this k-block (VALU, co-executes) ----
        #pragma unroll
        for (int r = 0; r < 2; ++r) {
            int kg = kb * 32 + kk0 + r;
            const float* wr = w1 + (size_t)kg * NQ;
            float s = 0.f;
            #pragma unroll
            for (int j = 0; j < NQ; ++j) s = fmaf(qreg[j], wr[j], s);
            h_lds[buf][m_h * 32 + kk0 + r] = f2bfu(fmaxf(s, 0.f));
        }
        __syncthreads();
    // ---- A fragment (contiguous k = 16*half + s, matches w2f): two 16B LDS reads ----
    const unsigned short* hb = h_lds[buf];
    v8bf lo = *(const v8bf*)(hb + ln * 32 + half * 16);
    v8bf hi = *(const v8bf*)(hb + ln * 32 + half * 16 + 8);
        v16bf a = __builtin_shufflevector(lo, hi, 0, 1, 2, 3, 4, 5, 6, 7,
                                          8, 9, 10, 11, 12, 13, 14, 15);
        #pragma unroll
        for (int t = 0; t < 8; ++t) {
            acc[t] = __builtin_amdgcn_wmma_f32_16x16x32_bf16(
                false, a, false, bfr[t], (short)0, acc[t], false, false);
        }
    }

    // ---- epilogue: residual add, LN row stats via LDS float atomics ----
    __syncthreads();
    #pragma unroll
    for (int t = 0; t < 8; ++t) {
        int e = (wave * 8 + t) * 16 + ln;
        #pragma unroll
        for (int r = 0; r < 8; ++r) {
            int m = half * 8 + r;
            acc[t][r] += x1[(size_t)(tok_base + m) * EMBED + e];
        }
    }
    #pragma unroll
    for (int r = 0; r < 8; ++r) {
        int m = half * 8 + r;
        float ps = 0.f, pss = 0.f;
        #pragma unroll
        for (int t = 0; t < 8; ++t) {
            float v = acc[t][r];
            ps += v;
            pss = fmaf(v, v, pss);
        }
        atomicAdd(&rowsum[m], ps);
        atomicAdd(&rowss[m], pss);
    }
    __syncthreads();
    if (tid < 16) {
        float mu = rowsum[tid] * (1.0f / EMBED);
        float var = rowss[tid] * (1.0f / EMBED) - mu * mu;
        mu_s[tid] = mu;
        rs_s[tid] = rsqrtf(var + LN_EPS);
    }
    __syncthreads();
    #pragma unroll
    for (int t = 0; t < 8; ++t) {
        int e = (wave * 8 + t) * 16 + ln;
        float ge = g2[e], be = b2[e];
        #pragma unroll
        for (int r = 0; r < 8; ++r) {
            int m = half * 8 + r;
            out[(size_t)(tok_base + m) * EMBED + e] =
                (acc[t][r] - mu_s[m]) * rs_s[m] * ge + be;
        }
    }
}

// ---------------------------------------------------------------------------
extern "C" void kernel_launch(void* const* d_in, const int* in_sizes, int n_in,
                              void* d_out, int out_size, void* d_ws, size_t ws_size,
                              hipStream_t stream) {
    const float* x   = (const float*)d_in[0];
    const float* tha = (const float*)d_in[1];
    const float* thf = (const float*)d_in[2];
    const float* wc  = (const float*)d_in[3];
    const float* w1  = (const float*)d_in[4];
    const float* w2  = (const float*)d_in[5];
    const float* g1  = (const float*)d_in[6];
    const float* b1  = (const float*)d_in[7];
    const float* g2  = (const float*)d_in[8];
    const float* b2  = (const float*)d_in[9];
    float* out = (float*)d_out;

    char* ws = (char*)d_ws;
    float*          qv   = (float*)(ws);                              // 1 MB
    float*          attn = (float*)(ws + (1u << 20));                 // 1 MB
    float*          x1   = (float*)(ws + (2u << 20));                 // 32 MB
    unsigned short* w2f  = (unsigned short*)(ws + (34u << 20));       // 8 MB

    k_prep_w2<<<(EMBED * FFN + 255) / 256, 256, 0, stream>>>(w2, w2f);
    k_qv<<<(BS * NQ + 255) / 256, 256, 0, stream>>>(x, tha, qv);
    k_attn<<<dim3(4, 128), 256, 0, stream>>>(qv, attn);
    k_x1<<<BS, 256, 0, stream>>>(x, attn, wc, g1, b1, x1);
    k_ffn<<<BS / 16, 256, 0, stream>>>(x1, thf, w1, w2f, g2, b2, out);
}
